// MultiHeadedAttention_7516192768966
// MI455X (gfx1250) — compile-verified, matrix operand order checked by reading, not run on hardware
//
#include <hip/hip_runtime.h>
#include <math.h>

// Problem constants (B=8, N=128, D=256, H=8, DK=32)
#define BB 8
#define NN 128
#define DD 256
#define HH 8
#define DKK 32

typedef float v2f  __attribute__((ext_vector_type(2)));
typedef float v8f  __attribute__((ext_vector_type(8)));
typedef __bf16 v16bf __attribute__((ext_vector_type(16)));
typedef unsigned short v8us __attribute__((ext_vector_type(8)));

union BFfrag { v16bf v; v8us h[2]; };

// RNE float -> bf16 split: f ~= hi + lo (both bf16), done in integer domain.
__device__ __forceinline__ void bf16_split(float f, unsigned short& hi, unsigned short& lo)
{
    unsigned int u = __float_as_uint(f);
    unsigned short h = (unsigned short)((u + 0x7FFFu + ((u >> 16) & 1u)) >> 16);
    float r = f - __uint_as_float(((unsigned int)h) << 16);
    unsigned int ur = __float_as_uint(r);
    lo = (unsigned short)((ur + 0x7FFFu + ((ur >> 16) & 1u)) >> 16);
    hi = h;
}

// ---------------------------------------------------------------------------
// Split weight matrix W[256,256] (f32) into bf16 hi/lo arrays (row-major).
// ---------------------------------------------------------------------------
__global__ __launch_bounds__(256)
void wsplit_kernel(const float* __restrict__ W,
                   unsigned short* __restrict__ Wh, unsigned short* __restrict__ Wl)
{
    const int i = blockIdx.x * 256 + threadIdx.x;   // < 65536
    unsigned short h, l;
    bf16_split(W[i], h, l);
    Wh[i] = h; Wl[i] = l;
}

// ---------------------------------------------------------------------------
// Split-bf16 WMMA GEMM: C[M,256] = A[M,256] @ W^T + bias, where W is given as
// pre-split bf16 hi/lo arrays.  a*w ~= ah*wh + ah*wl + al*wh  (3 bf16 WMMAs,
// ~1e-5 relative accuracy, 8/3x the f32-WMMA math rate).
// Block: 256 threads (8 wave32), tile 32 M-rows x 256 N-cols.
// Optionally scales A per-(row, head=col/32) by msgScale during staging.
// K=256 in 8 steps of V_WMMA_F32_16X16X32_BF16.
// ---------------------------------------------------------------------------
__global__ __launch_bounds__(256)
void gemm_bf16split(const float* __restrict__ A,
                    const unsigned short* __restrict__ Wh,
                    const unsigned short* __restrict__ Wl,
                    const float* __restrict__ bias, float* __restrict__ C,
                    const float* __restrict__ msgScale, int useScale)
{
    __shared__ __align__(16) unsigned short ldsAh[32 * 264];  // stride 264 (16B pad)
    __shared__ __align__(16) unsigned short ldsAl[32 * 264];

    const int t    = threadIdx.x;
    const int lane = t & 31;
    const int wave = t >> 5;
    const int m0   = blockIdx.x * 32;

    // ---- stage A tile 32x256, split into bf16 hi/lo ----
    {
        const int r  = t >> 3;                 // 0..31 local row
        const int c0 = (t & 7) * 32;           // 32-col chunk == one head block
        const int g  = m0 + r;                 // global row
        float s = 1.0f;
        if (useScale) {
            const int b = g >> 14, m = (g >> 7) & 127, n = g & 127, h = t & 7;
            s = msgScale[(((size_t)(b * 8 + h) * 128 + m) * 128) + n];
        }
        const float* src = A + (size_t)g * 256 + c0;
        unsigned short* dh = &ldsAh[r * 264 + c0];
        unsigned short* dl = &ldsAl[r * 264 + c0];
        #pragma unroll
        for (int i = 0; i < 8; ++i) {
            float4 x = *(const float4*)(src + i * 4);
            unsigned short h0, l0, h1, l1, h2, l2, h3, l3;
            bf16_split(x.x * s, h0, l0);
            bf16_split(x.y * s, h1, l1);
            bf16_split(x.z * s, h2, l2);
            bf16_split(x.w * s, h3, l3);
            dh[i*4+0] = h0; dh[i*4+1] = h1; dh[i*4+2] = h2; dh[i*4+3] = h3;
            dl[i*4+0] = l0; dl[i*4+1] = l1; dl[i*4+2] = l2; dl[i*4+3] = l3;
        }
    }
    __syncthreads();

    const int mi = wave >> 2;                  // 0..1 : 16-row half
    const int ni = wave & 3;                   // 0..3 : 64-col quarter
    const int lr = lane & 15;
    const int lh = lane >> 4;
    const int arow = mi * 16 + lr;

    v8f acc[4] = {{}, {}, {}, {}};

    #pragma unroll 2
    for (int k0 = 0; k0 < 256; k0 += 32) {
        // ISA 16-bit A layout: v0-3 = K[8*lh .. 8*lh+7], v4-7 = K[16+8*lh ..]
        const int kc0 = k0 + 8 * lh;
        const int kc1 = k0 + 16 + 8 * lh;
        BFfrag a_h, a_l;
        a_h.h[0] = *(const v8us*)&ldsAh[arow * 264 + kc0];
        a_h.h[1] = *(const v8us*)&ldsAh[arow * 264 + kc1];
        a_l.h[0] = *(const v8us*)&ldsAl[arow * 264 + kc0];
        a_l.h[1] = *(const v8us*)&ldsAl[arow * 264 + kc1];

        #pragma unroll
        for (int s2 = 0; s2 < 4; ++s2) {
            const int n = ni * 64 + s2 * 16 + lr;   // W row (B column)
            BFfrag b_h, b_l;
            b_h.h[0] = *(const v8us*)&Wh[(size_t)n * 256 + kc0];
            b_h.h[1] = *(const v8us*)&Wh[(size_t)n * 256 + kc1];
            b_l.h[0] = *(const v8us*)&Wl[(size_t)n * 256 + kc0];
            b_l.h[1] = *(const v8us*)&Wl[(size_t)n * 256 + kc1];
            acc[s2] = __builtin_amdgcn_wmma_f32_16x16x32_bf16(false, a_h.v, false, b_h.v,
                                                              (short)0, acc[s2], false, false);
            acc[s2] = __builtin_amdgcn_wmma_f32_16x16x32_bf16(false, a_h.v, false, b_l.v,
                                                              (short)0, acc[s2], false, false);
            acc[s2] = __builtin_amdgcn_wmma_f32_16x16x32_bf16(false, a_l.v, false, b_h.v,
                                                              (short)0, acc[s2], false, false);
        }
    }

    // ---- D layout: vgpr i -> row m0sub + i + 8*(lane>>4), col = lane&15 ----
    const int crow  = m0 + mi * 16 + 8 * lh;
    const int ccol0 = ni * 64 + lr;
    #pragma unroll
    for (int s2 = 0; s2 < 4; ++s2) {
        const int col = ccol0 + s2 * 16;
        const float bv = bias[col];
        #pragma unroll
        for (int i = 0; i < 8; ++i) {
            C[(size_t)(crow + i) * 256 + col] = acc[s2][i] + bv;
        }
    }
}

// ---------------------------------------------------------------------------
// Full-precision f32 WMMA GEMM (small matrices: q/v/node projections):
// C[M,256] = alpha * (A[M,256] @ W[256,256]^T + bias)
// ---------------------------------------------------------------------------
__global__ __launch_bounds__(256)
void gemm_wmma_f32(const float* __restrict__ A, const float* __restrict__ W,
                   const float* __restrict__ bias, float* __restrict__ C,
                   float alpha)
{
    __shared__ float ldsA[32 * 258];           // 32x256 A tile, stride 258 (pad)

    const int t    = threadIdx.x;
    const int lane = t & 31;
    const int wave = t >> 5;
    const int m0   = blockIdx.x * 32;

    {
        const int r  = t >> 3;
        const int c0 = (t & 7) * 32;
        const float* src = A + (size_t)(m0 + r) * 256 + c0;
        float* dst = &ldsA[r * 258 + c0];
        #pragma unroll
        for (int i = 0; i < 8; ++i) {
            float4 x = *(const float4*)(src + i * 4);
            dst[i*4+0] = x.x; dst[i*4+1] = x.y; dst[i*4+2] = x.z; dst[i*4+3] = x.w;
        }
    }
    __syncthreads();

    const int mi   = wave >> 2;
    const int ni   = wave & 3;
    const int arow = mi * 16 + (lane & 15);
    const int koff = 2 * (lane >> 4);
    const int nb   = ni * 64 + (lane & 15);

    v8f acc0 = {}, acc1 = {}, acc2 = {}, acc3 = {};

    #pragma unroll 4
    for (int k0 = 0; k0 < 256; k0 += 4) {
        v2f a  = *(const v2f*)&ldsA[arow * 258 + k0 + koff];
        v2f b0 = *(const v2f*)&W[(size_t)(nb +  0) * 256 + k0 + koff];
        v2f b1 = *(const v2f*)&W[(size_t)(nb + 16) * 256 + k0 + koff];
        v2f b2 = *(const v2f*)&W[(size_t)(nb + 32) * 256 + k0 + koff];
        v2f b3 = *(const v2f*)&W[(size_t)(nb + 48) * 256 + k0 + koff];
        acc0 = __builtin_amdgcn_wmma_f32_16x16x4_f32(false, a, false, b0, (short)0, acc0, false, false);
        acc1 = __builtin_amdgcn_wmma_f32_16x16x4_f32(false, a, false, b1, (short)0, acc1, false, false);
        acc2 = __builtin_amdgcn_wmma_f32_16x16x4_f32(false, a, false, b2, (short)0, acc2, false, false);
        acc3 = __builtin_amdgcn_wmma_f32_16x16x4_f32(false, a, false, b3, (short)0, acc3, false, false);
    }

    const int crow  = m0 + mi * 16 + 8 * (lane >> 4);
    const int ccol0 = ni * 64 + (lane & 15);
    v8f accs[4] = {acc0, acc1, acc2, acc3};
    #pragma unroll
    for (int s2 = 0; s2 < 4; ++s2) {
        const int col = ccol0 + s2 * 16;
        const float bv = bias[col];
        #pragma unroll
        for (int i = 0; i < 8; ++i) {
            C[(size_t)(crow + i) * 256 + col] = alpha * (accs[s2][i] + bv);
        }
    }
}

// ---------------------------------------------------------------------------
// Adjacency gating: adjS[b,i,:] = softmax_j( -lambda * adj[b,i,j] ), masked.
// ---------------------------------------------------------------------------
__global__ __launch_bounds__(128)
void adj_softmax_kernel(const float* __restrict__ adj, const int* __restrict__ mask,
                        const float* __restrict__ lam, float* __restrict__ out)
{
    __shared__ float red[128];
    const int bi = blockIdx.x;                 // b*128 + i
    const int b  = bi >> 7;
    const int j  = threadIdx.x;
    const float lambda = lam[0];

    float x = -lambda * adj[(size_t)bi * 128 + j];
    if (mask[b * 128 + j] == 0) x = -__builtin_inff();

    red[j] = x; __syncthreads();
    for (int s = 64; s > 0; s >>= 1) { if (j < s) red[j] = fmaxf(red[j], red[j + s]); __syncthreads(); }
    const float mx = red[0]; __syncthreads();

    const float e = __expf(x - mx);
    red[j] = e; __syncthreads();
    for (int s = 64; s > 0; s >>= 1) { if (j < s) red[j] += red[j + s]; __syncthreads(); }
    out[(size_t)bi * 128 + j] = e / red[0];
}

// ---------------------------------------------------------------------------
// Scores: block per (b,m); thread n owns edge (b,m,n).
// ---------------------------------------------------------------------------
__global__ __launch_bounds__(128)
void scores_kernel(const float* __restrict__ kproj, const float* __restrict__ qW,
                   const int* __restrict__ mask,
                   float* __restrict__ outS, float* __restrict__ inS)
{
    const int bm = blockIdx.x;                 // b*128 + m
    const int b  = bm >> 7, m = bm & 127;
    const int n  = threadIdx.x;

    const float* krow = kproj + ((size_t)bm * 128 + n) * 256;
    const float* qm   = qW + (size_t)bm * 256;
    const float* qn   = qW + (size_t)(b * 128 + n) * 256;

    const bool mn0 = (mask[b * 128 + n] == 0);
    const bool mm0 = (mask[b * 128 + m] == 0);

    #pragma unroll
    for (int h = 0; h < 8; ++h) {
        float so = 0.f, si = 0.f;
        #pragma unroll
        for (int d4 = 0; d4 < 8; ++d4) {
            const int off = h * 32 + d4 * 4;
            float4 kv  = *(const float4*)(krow + off);
            float4 qmv = *(const float4*)(qm + off);
            float4 qnv = *(const float4*)(qn + off);
            so += kv.x * qmv.x + kv.y * qmv.y + kv.z * qmv.z + kv.w * qmv.w;
            si += kv.x * qnv.x + kv.y * qnv.y + kv.z * qnv.z + kv.w * qnv.w;
        }
        const size_t ob = (size_t)(b * 8 + h) * 128;
        outS[(ob + m) * 128 + n] = mn0 ? -__builtin_inff() : so;
        inS [(ob + n) * 128 + m] = mm0 ? -__builtin_inff() : si;
    }
}

// ---------------------------------------------------------------------------
// In-place softmax over rows of 128 (covers outS and inS back-to-back).
// ---------------------------------------------------------------------------
__global__ __launch_bounds__(128)
void row_softmax_kernel(float* __restrict__ buf)
{
    __shared__ float red[128];
    const size_t row = blockIdx.x;
    const int j = threadIdx.x;

    float x = buf[row * 128 + j];
    red[j] = x; __syncthreads();
    for (int s = 64; s > 0; s >>= 1) { if (j < s) red[j] = fmaxf(red[j], red[j + s]); __syncthreads(); }
    const float mx = red[0]; __syncthreads();

    const float e = __expf(x - mx);
    red[j] = e; __syncthreads();
    for (int s = 64; s > 0; s >>= 1) { if (j < s) red[j] += red[j + s]; __syncthreads(); }
    buf[row * 128 + j] = e / red[0];
}

// ---------------------------------------------------------------------------
// message[b,h,i,j] = (i==j ? in_attn : out_attn + in_attn) * adjS[b,i,j]
// ---------------------------------------------------------------------------
__global__ __launch_bounds__(256)
void message_kernel(const float* __restrict__ oa, const float* __restrict__ ia,
                    const float* __restrict__ adjS, float* __restrict__ msg)
{
    const int tid = blockIdx.x * 256 + threadIdx.x;   // < B*H*N*N = 1048576
    const int j = tid & 127, i = (tid >> 7) & 127, b = tid >> 17;
    const float o = oa[tid], v = ia[tid];
    const float r = (i == j) ? v : (o + v);
    msg[tid] = r * adjS[((size_t)(b * 128 + i) * 128) + j];
}

// ---------------------------------------------------------------------------
// node_hidden[b,m, h*32+d] = sum_n msg[b,h,m,n] * v[b,n, h*32+d]
// ---------------------------------------------------------------------------
__global__ __launch_bounds__(256)
void node_hidden_kernel(const float* __restrict__ msg, const float* __restrict__ vW,
                        float* __restrict__ nh)
{
    const int bm = blockIdx.x;
    const int b = bm >> 7, m = bm & 127;
    const int t = threadIdx.x;
    const int h = t >> 5;

    const float* mrow  = msg + ((size_t)(b * 8 + h) * 128 + m) * 128;
    const float* vbase = vW + (size_t)b * 128 * 256 + t;
    float acc = 0.f;
    #pragma unroll 4
    for (int n = 0; n < 128; ++n) acc += mrow[n] * vbase[(size_t)n * 256];
    nh[(size_t)bm * 256 + t] = acc;
}

// ---------------------------------------------------------------------------
extern "C" void kernel_launch(void* const* d_in, const int* in_sizes, int n_in,
                              void* d_out, int out_size, void* d_ws, size_t ws_size,
                              hipStream_t stream)
{
    const float* query    = (const float*)d_in[0];
    const float* value    = (const float*)d_in[1];
    const float* key_edge = (const float*)d_in[2];
    const float* adj      = (const float*)d_in[3];
    const int*   mask     = (const int*)  d_in[4];
    const float* Wq  = (const float*)d_in[5];
    const float* bq  = (const float*)d_in[6];
    const float* Wk  = (const float*)d_in[7];
    const float* bk  = (const float*)d_in[8];
    const float* Wv  = (const float*)d_in[9];
    const float* bv  = (const float*)d_in[10];
    const float* Wno = (const float*)d_in[11];
    const float* bno = (const float*)d_in[12];
    const float* Weo = (const float*)d_in[13];
    const float* beo = (const float*)d_in[14];
    const float* lam = (const float*)d_in[15];

    float* out = (float*)d_out;
    float* ws  = (float*)d_ws;

    // workspace layout (floats): ~16.3 MB + 512 KB bf16 weights
    float* adjS = ws;                  // 131072
    float* qW   = adjS + 131072;       // 262144
    float* vW   = qW   + 262144;       // 262144
    float* outS = vW   + 262144;       // 1048576
    float* inS  = outS + 1048576;      // 1048576 (contiguous after outS)
    float* msg  = inS  + 1048576;      // 1048576
    float* nh   = msg  + 1048576;      // 262144
    unsigned short* WkH = (unsigned short*)(nh + 262144);  // 65536 each
    unsigned short* WkL = WkH + 65536;
    unsigned short* WeH = WkL + 65536;
    unsigned short* WeL = WeH + 65536;

    float* nodeOut = out;              // [B,N,D] = 262144
    float* kproj   = out + 262144;     // k_proj in (and overwritten in place
                                       // by) the edge-output region of d_out

    const float inv_sqrt_dk = 0.17677669529663687f;   // 1/sqrt(32)

    // 0) split big-GEMM weights into bf16 hi/lo (L2-resident, 512 KB)
    wsplit_kernel<<<256, 256, 0, stream>>>(Wk,  WkH, WkL);
    wsplit_kernel<<<256, 256, 0, stream>>>(Weo, WeH, WeL);
    // 1) adjacency gating softmax
    adj_softmax_kernel<<<BB * NN, 128, 0, stream>>>(adj, mask, lam, adjS);
    // 2) q/v projections (exact f32 WMMA; q pre-scaled by 1/sqrt(DK))
    gemm_wmma_f32<<<32, 256, 0, stream>>>(query, Wq, bq, qW, inv_sqrt_dk);
    gemm_wmma_f32<<<32, 256, 0, stream>>>(value, Wv, bv, vW, 1.0f);
    // 3) big GEMM: k = key_edge @ Wk^T + bk (split-bf16, into edge region)
    gemm_bf16split<<<4096, 256, 0, stream>>>(key_edge, WkH, WkL, bk, kproj, nullptr, 0);
    // 4) out/in attention scores
    scores_kernel<<<BB * NN, 128, 0, stream>>>(kproj, qW, mask, outS, inS);
    // 5) softmax both score tensors in place (contiguous rows)
    row_softmax_kernel<<<2 * BB * HH * NN, 128, 0, stream>>>(outS);
    // 6) gated message
    message_kernel<<<4096, 256, 0, stream>>>(outS, inS, adjS, msg);
    // 7) node aggregation + node output projection (exact f32 WMMA)
    node_hidden_kernel<<<BB * NN, 256, 0, stream>>>(msg, vW, nh);
    gemm_wmma_f32<<<32, 256, 0, stream>>>(nh, Wno, bno, nodeOut, 1.0f);
    // 8) edge output: (message ⊙ k_proj) @ Weo^T + beo (split-bf16, in place)
    gemm_bf16split<<<4096, 256, 0, stream>>>(kproj, WeH, WeL, beo, kproj, msg, 1);

    (void)in_sizes; (void)n_in; (void)out_size; (void)ws_size;
}
